// FilterDetections_80788334837794
// MI455X (gfx1250) — compile-verified, matrix operand order checked by reading, not run on hardware
//
#include <hip/hip_runtime.h>
#include <stdint.h>

#define SCORE_T 0.05f
#define NMS_THR 0.5f
#define NEGS    (-1e9f)
#define KSEL    500        // per-class candidate count (reference K)
#define KPAD    512        // padded to power of two for bitonic sort
#define NBINS   2048
#define MAXDET  300

typedef float    v2f  __attribute__((ext_vector_type(2)));
typedef float    v8f  __attribute__((ext_vector_type(8)));
typedef _Float16 v16h __attribute__((ext_vector_type(16)));

#if __has_builtin(__builtin_amdgcn_ballot_w32)
#define WAVE_BALLOT(p) __builtin_amdgcn_ballot_w32(p)
#else
#define WAVE_BALLOT(p) ((unsigned)__ballot(p))
#endif

// --- CDNA5 async global->LDS copies (tracked by ASYNCcnt) --------------------
// LDS byte address = low 32 bits of the generic pointer (LDS aperture rule).
__device__ __forceinline__ void async_g2l_b32(void* lds, const void* gaddr) {
    unsigned l = (unsigned)(uintptr_t)lds;
    asm volatile("global_load_async_to_lds_b32 %0, %1, off"
                 :: "v"(l), "v"(gaddr) : "memory");
}
__device__ __forceinline__ void async_g2l_b128(void* lds, const void* gaddr) {
    unsigned l = (unsigned)(uintptr_t)lds;
    asm volatile("global_load_async_to_lds_b128 %0, %1, off"
                 :: "v"(l), "v"(gaddr) : "memory");
}
__device__ __forceinline__ void async_wait0() {
    asm volatile("s_wait_asynccnt 0x0" ::: "memory");
}

// Monotonic bin for positive floats in (SCORE_T, 1.0]: top bits of the IEEE pattern.
__device__ __forceinline__ int score_bin(float s) {
    unsigned u = __float_as_uint(s);
    int b = (int)(u >> 15) - 30720;      // 0.05 -> ~665, 1.0 -> 1792
    return b < 0 ? 0 : (b > NBINS - 1 ? NBINS - 1 : b);
}

// ---------------------------------------------------------------------------
// Kernel 0: init workspace
// ---------------------------------------------------------------------------
__global__ void k_init(unsigned* __restrict__ hist, int* __restrict__ cntA,
                       int* __restrict__ cntB, float* __restrict__ candS,
                       int* __restrict__ candI, float* __restrict__ nmsS, int pairs) {
    int i = blockIdx.x * blockDim.x + threadIdx.x;
    if (i < pairs * NBINS) hist[i] = 0u;
    if (i < pairs * KPAD) { candS[i] = NEGS; candI[i] = -1; nmsS[i] = NEGS; }
    if (i < pairs)        { cntA[i] = 0; cntB[i] = 0; }
}

// ---------------------------------------------------------------------------
// Kernel 1: per-(image,class) score histogram (streaming pass over 128 MB)
// ---------------------------------------------------------------------------
__global__ void k_hist(const float* __restrict__ cls, unsigned* __restrict__ hist,
                       int total, int C, int NC) {
    int i = blockIdx.x * blockDim.x + threadIdx.x;
    if (i >= total) return;
    if (i + 16384 < total) __builtin_prefetch(cls + i + 16384, 0, 0); // global_prefetch_b8
    float s = cls[i];
    if (s > SCORE_T) {
        int b = i / NC;
        int c = i % C;
        atomicAdd(&hist[(b * C + c) * NBINS + score_bin(s)], 1u);
    }
}

// ---------------------------------------------------------------------------
// Kernel 2: suffix-scan each pair's histogram to find the K-th score bin
// ---------------------------------------------------------------------------
__global__ void k_scan(const unsigned* __restrict__ hist, int* __restrict__ tbin,
                       int* __restrict__ nAb, int* __restrict__ rem, int pairs) {
    int p = blockIdx.x * blockDim.x + threadIdx.x;
    if (p >= pairs) return;
    const unsigned* h = hist + (size_t)p * NBINS;
    int cum = 0, tb = -1, na = 0, rm = 0;
    for (int bin = NBINS - 1; bin >= 0; --bin) {
        int c = (int)h[bin];
        if (cum + c >= KSEL) { tb = bin; na = cum; rm = KSEL - cum; break; }
        cum += c;
    }
    if (tb < 0) { na = cum; rm = 0; }   // fewer than K valid scores
    tbin[p] = tb; nAb[p] = na; rem[p] = rm;
}

// ---------------------------------------------------------------------------
// Kernel 3: collect top-K candidates (second streaming pass)
// ---------------------------------------------------------------------------
__global__ void k_collect(const float* __restrict__ cls, const int* __restrict__ tbin,
                          const int* __restrict__ nAb, const int* __restrict__ rem,
                          int* __restrict__ cntA, int* __restrict__ cntB,
                          float* __restrict__ candS, int* __restrict__ candI,
                          int total, int C, int NC) {
    int i = blockIdx.x * blockDim.x + threadIdx.x;
    if (i >= total) return;
    float s = cls[i];
    if (s <= SCORE_T) return;
    int b = i / NC;
    int c = i % C;
    int n = (i % NC) / C;
    int pr = b * C + c;
    int bin = score_bin(s);
    int tb = tbin[pr];
    if (bin > tb) {
        int pos = atomicAdd(&cntA[pr], 1);
        candS[pr * KPAD + pos] = s; candI[pr * KPAD + pos] = n;
    } else if (bin == tb) {
        int r = atomicAdd(&cntB[pr], 1);
        if (r < rem[pr]) {
            int pos = nAb[pr] + r;
            candS[pr * KPAD + pos] = s; candI[pr * KPAD + pos] = n;
        }
    }
}

// ---------------------------------------------------------------------------
// LDS bitonic sort, 512 elements, descending by key (512 threads)
// ---------------------------------------------------------------------------
__device__ __forceinline__ void bitonic512_desc(float* key, int* val) {
    int t = threadIdx.x;
    for (int k = 2; k <= KPAD; k <<= 1) {
        for (int j = k >> 1; j > 0; j >>= 1) {
            int p = t ^ j;
            if (p > t) {
                bool dir = ((t & k) == 0);
                float ka = key[t], kb = key[p];
                if ((ka < kb) == dir) {
                    key[t] = kb; key[p] = ka;
                    int va = val[t]; val[t] = val[p]; val[p] = va;
                }
            }
            __syncthreads();
        }
    }
}

// ---------------------------------------------------------------------------
// areasum 16x16 tile via WMMA: D[M][N] = area[r0+M] + area[c0+N]
// A (16x4 f32) rows = (area_i, 1, 0, 0); B (4x16 f32) cols = (1, area_j, 0, 0)
// ---------------------------------------------------------------------------
__device__ __forceinline__ v8f areasum_tile(const float* sarea, int r0, int c0, int lane) {
#if __has_builtin(__builtin_amdgcn_wmma_f32_16x16x4_f32)
    v2f a = {0.f, 0.f}, b = {0.f, 0.f};
    if (lane < 16) {
        a.x = sarea[r0 + lane]; a.y = 1.0f;      // K=0 row area, K=1 ones (lanes 16-31: K=2,3 = 0)
        b.x = 1.0f;             b.y = sarea[c0 + lane]; // B K=0 ones, K=1 areas
    }
    v8f c = {};
    return __builtin_amdgcn_wmma_f32_16x16x4_f32(false, a, false, b, (short)0, c, false, false);
#elif __has_builtin(__builtin_amdgcn_wmma_f32_16x16x32_f16)
    v16h a = {}, b = {};
    if (lane < 16) {
        a[0] = (_Float16)sarea[r0 + lane]; a[1] = (_Float16)1.0f;
        b[0] = (_Float16)1.0f;             b[1] = (_Float16)sarea[c0 + lane];
    }
    v8f c = {};
    return __builtin_amdgcn_wmma_f32_16x16x32_f16(false, a, false, b, (short)0, c, false, false);
#else
    v8f d;
    for (int v = 0; v < 8; ++v) {
        int i = r0 + v + ((lane >> 4) << 3);
        int j = c0 + (lane & 15);
        d[v] = sarea[i] + sarea[j];
    }
    return d;
#endif
}

// ---------------------------------------------------------------------------
// Kernel 4: per-(image,class) sort + suppression-matrix + greedy NMS + cap
// one block of 512 threads per pair
// ---------------------------------------------------------------------------
__global__ __launch_bounds__(512)
void k_sortnms(const float4* __restrict__ boxes, const float* __restrict__ candS,
               int* __restrict__ candI, float* __restrict__ nmsS, int N, int C) {
    __shared__ float    skey[KPAD];
    __shared__ int      sidx[KPAD];
    __shared__ float4   sbox[KPAD];
    __shared__ float    sarea[KPAD];
    __shared__ unsigned ssup[KPAD * 16];   // 512 rows x 512 cols bitmask = 32 KB
    __shared__ unsigned skeep[16];

    int pair = blockIdx.x;
    int tid  = threadIdx.x;
    int lane = tid & 31;
    int wv   = tid >> 5;              // 16 waves
    int b    = pair / C;

    // stage candidates into LDS with CDNA5 async copies (ASYNCcnt path)
    async_g2l_b32(&skey[tid], &candS[(size_t)pair * KPAD + tid]);
    async_g2l_b32(&sidx[tid], &candI[(size_t)pair * KPAD + tid]);
    async_wait0();
    __syncthreads();

    bitonic512_desc(skey, sidx);      // descending; invalid (NEGS) sink to the end

    // gather boxes for sorted candidates; degenerate (zero) box for invalid slots
    int  n     = sidx[tid];
    bool valid = (skey[tid] > SCORE_T) && (n >= 0);
    sbox[tid]  = make_float4(0.f, 0.f, 0.f, 0.f);
    __syncthreads();                  // zero-fill visible before async overwrite
    if (valid) async_g2l_b128(&sbox[tid], &boxes[b * N + n]);
    async_wait0();
    __syncthreads();

    {
        float4 bx = sbox[tid];
        sarea[tid] = (bx.z - bx.x) * (bx.w - bx.y);
    }
    for (int k = tid; k < KPAD * 16; k += 512) ssup[k] = 0u;
    __syncthreads();

    // build suppression bitmask: 32x32 tiles of 16x16, WMMA for the areasum
    for (int t = wv; t < 32 * 32; t += 16) {
        int r0 = (t >> 5) * 16;
        int c0 = (t & 31) * 16;
        v8f asum = areasum_tile(sarea, r0, c0, lane);
        #pragma unroll
        for (int v = 0; v < 8; ++v) {
            // C/D layout: lanes 0-15 -> M=v, lanes 16-31 -> M=v+8; N = lane%16
            int i = r0 + v + ((lane >> 4) << 3);
            int j = c0 + (lane & 15);
            float4 bi = sbox[i], bj = sbox[j];
            float lx = fmaxf(bi.x, bj.x), ly = fmaxf(bi.y, bj.y);
            float rx = fminf(bi.z, bj.z), ry = fminf(bi.w, bj.w);
            float w  = fmaxf(rx - lx, 0.f), h = fmaxf(ry - ly, 0.f);
            float inter = w * h;
            // iou > T  <=>  inter > T*(areasum - inter + eps)   (denominator > 0)
            bool sup_p = inter > NMS_THR * (asum[v] - inter + 1e-9f);
            unsigned m = WAVE_BALLOT(sup_p);
            if (lane == 0) {
                unsigned lo = m & 0xFFFFu, hi = m >> 16;
                int word = c0 >> 5, sh = c0 & 16;
                atomicOr(&ssup[(r0 + v)     * 16 + word], lo << sh);
                atomicOr(&ssup[(r0 + v + 8) * 16 + word], hi << sh);
            }
        }
    }
    __syncthreads();

    // greedy NMS scan on wave 0: lane w holds keep-word w (columns w*32..w*32+31)
    if (tid < 32) {
        unsigned keepw = 0u;
        if (lane < 16) {
            for (int bit = 0; bit < 32; ++bit)
                if (skey[lane * 32 + bit] > SCORE_T) keepw |= 1u << bit;
        }
        for (int i = 0; i < KPAD; ++i) {
            int w = i >> 5;
            unsigned kw = (unsigned)__shfl((int)keepw, w, 32);
            if ((kw >> (i & 31)) & 1u) {               // box i still alive
                unsigned row = (lane < 16) ? ssup[i * 16 + lane] : 0u;
                if (lane < w)       row = 0u;          // only suppress j > i
                else if (lane == w) row &= ((i & 31) == 31) ? 0u
                                         : (0xFFFFFFFFu << ((i & 31) + 1));
                keepw &= ~row;
            }
        }
        if (lane < 16) skeep[lane] = keepw;
    }
    __syncthreads();

    // cap per-class detections at MAXDET (rank < 300 over kept)
    if (tid == 0) {
        int taken = 0;
        for (int w = 0; w < 16; ++w) {
            unsigned m = skeep[w], out = 0u;
            for (int bit = 0; bit < 32; ++bit)
                if ((m >> bit) & 1u) { if (taken < MAXDET) { out |= 1u << bit; ++taken; } }
            skeep[w] = out;
        }
    }
    __syncthreads();

    bool kept = (skeep[tid >> 5] >> (tid & 31)) & 1u;
    float s = skey[tid];
    nmsS[(size_t)pair * KPAD + tid] = (kept && s > SCORE_T) ? s : NEGS;
    candI[(size_t)pair * KPAD + tid] = sidx[tid];   // store sorted indices
}

// ---------------------------------------------------------------------------
// Kernel 5: per-image global top-300 over C*512 surviving candidates
// ---------------------------------------------------------------------------
__global__ __launch_bounds__(512)
void k_final(const float4* __restrict__ boxes, const float* __restrict__ nmsS,
             const int* __restrict__ candI, float* __restrict__ outB,
             float* __restrict__ outS, int* __restrict__ outL, int N, int C) {
    __shared__ unsigned hist[NBINS];
    __shared__ float lkey[KPAD];
    __shared__ int   lpos[KPAD];
    __shared__ int   sh_tb, sh_na, sh_rem, sh_cA, sh_cB;

    int b   = blockIdx.x;
    int tid = threadIdx.x;
    int total = C * KPAD;

    for (int k = tid; k < NBINS; k += 512) hist[k] = 0u;
    lkey[tid] = NEGS; lpos[tid] = -1;
    if (tid == 0) { sh_cA = 0; sh_cB = 0; }
    __syncthreads();

    for (int p = tid; p < total; p += 512) {
        float s = nmsS[(size_t)b * total + p];
        if (s > SCORE_T) atomicAdd(&hist[score_bin(s)], 1u);
    }
    __syncthreads();

    if (tid == 0) {
        int cum = 0, tb = -1, na = 0, rm = 0;
        for (int bin = NBINS - 1; bin >= 0; --bin) {
            int c = (int)hist[bin];
            if (cum + c >= MAXDET) { tb = bin; na = cum; rm = MAXDET - cum; break; }
            cum += c;
        }
        if (tb < 0) { na = cum; rm = 0; }
        sh_tb = tb; sh_na = na; sh_rem = rm;
    }
    __syncthreads();

    for (int p = tid; p < total; p += 512) {
        float s = nmsS[(size_t)b * total + p];
        if (s > SCORE_T) {
            int bin = score_bin(s);
            if (bin > sh_tb) {
                int pos = atomicAdd(&sh_cA, 1);
                lkey[pos] = s; lpos[pos] = p;
            } else if (bin == sh_tb) {
                int r = atomicAdd(&sh_cB, 1);
                if (r < sh_rem) { int pos = sh_na + r; lkey[pos] = s; lpos[pos] = p; }
            }
        }
    }
    __syncthreads();

    bitonic512_desc(lkey, lpos);

    if (tid < MAXDET) {
        float s = lkey[tid];
        int p = lpos[tid];
        bool valid = (s > SCORE_T) && (p >= 0);
        int n = valid ? candI[(size_t)b * total + p] : -1;
        valid = valid && (n >= 0);
        float4 bx = make_float4(0.f, 0.f, 0.f, 0.f);
        if (valid) bx = boxes[b * N + n];
        int o = b * MAXDET + tid;
        outB[o * 4 + 0] = bx.x; outB[o * 4 + 1] = bx.y;
        outB[o * 4 + 2] = bx.z; outB[o * 4 + 3] = bx.w;
        outS[o] = valid ? s : 0.f;
        outL[o] = valid ? (p >> 9) : -1;     // p>>9 == class id (512 slots/class)
    }
}

// ---------------------------------------------------------------------------
extern "C" void kernel_launch(void* const* d_in, const int* in_sizes, int n_in,
                              void* d_out, int out_size, void* d_ws, size_t ws_size,
                              hipStream_t stream) {
    const float* boxes = (const float*)d_in[0];
    const float* cls   = (const float*)d_in[1];

    int B = out_size / (MAXDET * 6);             // 4*300 + 300 + 300 floats/ints per image
    int N = in_sizes[0] / (4 * B);
    int C = in_sizes[1] / (B * N);
    int pairs = B * C;
    int NC = N * C;
    int total = in_sizes[1];

    char* base = (char*)d_ws;
    unsigned* hist = (unsigned*)base; base += (size_t)pairs * NBINS * 4;
    int*   cntA  = (int*)base;   base += (size_t)pairs * 4;
    int*   cntB  = (int*)base;   base += (size_t)pairs * 4;
    int*   tbin  = (int*)base;   base += (size_t)pairs * 4;
    int*   nAb   = (int*)base;   base += (size_t)pairs * 4;
    int*   rem   = (int*)base;   base += (size_t)pairs * 4;
    float* candS = (float*)base; base += (size_t)pairs * KPAD * 4;
    int*   candI = (int*)base;   base += (size_t)pairs * KPAD * 4;
    float* nmsS  = (float*)base; base += (size_t)pairs * KPAD * 4;

    float* outB = (float*)d_out;
    float* outS = outB + (size_t)B * MAXDET * 4;
    int*   outL = (int*)(outS + (size_t)B * MAXDET);

    int initN = pairs * NBINS;
    k_init<<<(initN + 255) / 256, 256, 0, stream>>>(hist, cntA, cntB, candS, candI, nmsS, pairs);
    k_hist<<<(total + 255) / 256, 256, 0, stream>>>(cls, hist, total, C, NC);
    k_scan<<<(pairs + 63) / 64, 64, 0, stream>>>(hist, tbin, nAb, rem, pairs);
    k_collect<<<(total + 255) / 256, 256, 0, stream>>>(cls, tbin, nAb, rem, cntA, cntB,
                                                       candS, candI, total, C, NC);
    k_sortnms<<<pairs, 512, 0, stream>>>((const float4*)boxes, candS, candI, nmsS, N, C);
    k_final<<<B, 512, 0, stream>>>((const float4*)boxes, nmsS, candI, outB, outS, outL, N, C);
}
